// Transformer_Block_64287070486550
// MI455X (gfx1250) — matrix operand order repaired and checked by reading, not compiled, not run on hardware
//
#include <hip/hip_runtime.h>
#include <hip/hip_bf16.h>

// ---------------------------------------------------------------------------
// MI455X (gfx1250) transformer block. bf16 WMMA everywhere, fp32 accumulate.
// GEMM tile staging uses the CDNA5 Tensor Data Mover (tensor_load_to_lds,
// TENSORcnt) with LDS double buffering; falls back to sync loads if the
// builtin is unavailable.
// ---------------------------------------------------------------------------

typedef unsigned short u16;
typedef __attribute__((ext_vector_type(8)))  unsigned short us8;
typedef __attribute__((ext_vector_type(16))) unsigned short us16;
typedef __attribute__((ext_vector_type(16))) __bf16        v16bf;
typedef __attribute__((ext_vector_type(8)))  float         v8f;

__device__ __forceinline__ u16 f2bf(float f) {
  unsigned u = __float_as_uint(f);
  u += 0x7fffu + ((u >> 16) & 1u);          // round-to-nearest-even
  return (u16)(u >> 16);
}
__device__ __forceinline__ float bf2f(u16 h) {
  return __uint_as_float(((unsigned)h) << 16);
}
__device__ __forceinline__ v8f v8f_zero() {
  v8f z = {0.f, 0.f, 0.f, 0.f, 0.f, 0.f, 0.f, 0.f};
  return z;
}
__device__ __forceinline__ v8f wmma_bf16(us16 a, us16 b, v8f c) {
  return __builtin_amdgcn_wmma_f32_16x16x32_bf16(
      false, __builtin_bit_cast(v16bf, a),
      false, __builtin_bit_cast(v16bf, b),
      (short)0, c, false, false);
}

// ---------------------------------------------------------------------------
// Tensor Data Mover support (CDNA5 TDM). Guarded so absence cannot break
// the build; arity discriminated per bridge doc (5-arg ROCm7.2 / 6-arg
// therock which ships the TDM header).
// ---------------------------------------------------------------------------
#if defined(__has_builtin)
#if __has_builtin(__builtin_amdgcn_tensor_load_to_lds)
#define HAVE_TDM 1
#endif
#endif

#if defined(__has_include)
#if __has_include(<hip/amd_detail/amd_gfx1250_TDM.h>)
#define TDM_ARITY6 1
#endif
#endif

__device__ __forceinline__ void wait_tensorcnt0() {
#if defined(__has_builtin) && __has_builtin(__builtin_amdgcn_s_wait_tensorcnt)
  __builtin_amdgcn_s_wait_tensorcnt(0);
#else
  asm volatile("s_wait_tensorcnt 0x0" ::: "memory");
#endif
}

#if defined(HAVE_TDM)
typedef __attribute__((ext_vector_type(4))) unsigned int u32x4;
typedef __attribute__((ext_vector_type(8))) int          i32x8;
typedef __attribute__((ext_vector_type(4))) int          i32x4;

// low 32 bits of a generic pointer into LDS == LDS byte offset (AS3 is 32-bit)
__device__ __forceinline__ unsigned lds_off(const void* p) {
  return (unsigned)(unsigned long long)(uintptr_t)p;
}

// 2D tile DMA: global (rows x strideElems, bf16, row-major) -> LDS, tile of
// tileRows x tileK elements stored contiguously (row-major, row length tileK).
// D# layout per CDNA5 ISA sec 8.3/8.4 (group0 128b, group1 256b, groups 2/3 zero).
__device__ __forceinline__ void tdm_load_tile_2d(const u16* gsrc,
                                                 unsigned ldsByteOff,
                                                 int strideElems, int tileK,
                                                 int tileRows) {
  unsigned long long ga = (unsigned long long)(uintptr_t)gsrc;
  u32x4 g0;
  g0[0] = 1u;                                   // count=1, user mode, no gather
  g0[1] = ldsByteOff;                           // lds_addr (bytes)   [63:32]
  g0[2] = (unsigned)(ga & 0xffffffffu);         // global_addr[31:0]  [95:64]
  g0[3] = (unsigned)((ga >> 32) & 0x01ffffffu)  // global_addr[56:32] [120:96]
          | (2u << 30);                         // type=2 ("image")   [127:126]
  i32x8 g1;
  unsigned td0 = (unsigned)tileK;               // tensor_dim0 == tile_dim0
  unsigned td1 = (unsigned)tileRows;            // tensor_dim1 == tile_dim1
  unsigned st0 = (unsigned)strideElems;         // tensor_dim0_stride (elems)
  g1[0] = (int)(1u << 16);                      // wg_mask=0, data_size=1 (2B)
  g1[1] = (int)((td0 & 0xffffu) << 16);         // dim0[15:0]  at bits 63:48
  g1[2] = (int)((td0 >> 16) & 0xffffu)          // dim0[31:16] at bits 79:64
          | (int)((td1 & 0xffffu) << 16);       // dim1[15:0]  at bits 95:80
  g1[3] = (int)((td1 >> 16) & 0xffffu)          // dim1[31:16]
          | (int)(((unsigned)tileK & 0xffffu) << 16);   // tile_dim0 [127:112]
  g1[4] = (int)((unsigned)tileRows & 0xffffu);  // tile_dim1 [143:128], tile_dim2=0
  g1[5] = (int)st0;                             // dim0_stride[31:0] [191:160]
  g1[6] = 0;                                    // dim0_stride[47:32], dim1_stride lo
  g1[7] = 0;                                    // dim1_stride hi
  i32x4 z4 = {0, 0, 0, 0};
#if defined(TDM_ARITY6)
  i32x8 z8 = {0, 0, 0, 0, 0, 0, 0, 0};
  __builtin_amdgcn_tensor_load_to_lds(g0, g1, z4, z4, z8, 0);
#else
  __builtin_amdgcn_tensor_load_to_lds(g0, g1, z4, z4, 0);
#endif
}
#endif  // HAVE_TDM

// ---------------------------------------------------------------------------
// Elementwise helpers
// ---------------------------------------------------------------------------
__global__ void cast_bf16_kernel(const float* __restrict__ in,
                                 u16* __restrict__ out, size_t n) {
  size_t i = (size_t)blockIdx.x * blockDim.x + threadIdx.x;
  if (i < n) out[i] = f2bf(in[i]);
}

// w: [G, Kd, Nd] row-major fp32  ->  wt: [G*Nd, Kd] bf16 (i.e. W^T per group)
__global__ void transpose_w_kernel(const float* __restrict__ w,
                                   u16* __restrict__ wt,
                                   int G, int Kd, int Nd) {
  size_t i = (size_t)blockIdx.x * blockDim.x + threadIdx.x;
  size_t total = (size_t)G * Kd * Nd;
  if (i >= total) return;
  int n = (int)(i % Nd);
  size_t r = i / Nd;
  int k = (int)(r % Kd);
  int g = (int)(r / Kd);
  wt[((size_t)g * Nd + n) * Kd + k] = f2bf(w[i]);
}

// v: [B,S,H,DH] bf16 -> vt: [B,H,DH,S] bf16
__global__ void transpose_v_kernel(const u16* __restrict__ v,
                                   u16* __restrict__ vt,
                                   int Bn, int S, int Hn, int DHn) {
  size_t i = (size_t)blockIdx.x * blockDim.x + threadIdx.x;
  size_t total = (size_t)Bn * S * Hn * DHn;
  if (i >= total) return;
  int d = (int)(i % DHn);
  size_t r = i / DHn;
  int h = (int)(r % Hn);
  r /= Hn;
  int s = (int)(r % S);
  int b = (int)(r / S);
  vt[(((size_t)b * Hn + h) * DHn + d) * S + s] = v[i];
}

// ---------------------------------------------------------------------------
// bf16 WMMA GEMM:  C[M,N] = A[M,K] * Bt[N,K]^T  (+bias +resid, relu, f32/bf16)
// Block tile 128x128, K-step 32, 8 waves (4x2), wave tile 32x64.
// Tile staging: TDM double buffering (wave 0 issues DMA for tile i+1 while
// all waves run WMMAs on tile i; TENSORcnt + workgroup barrier to flip).
// ---------------------------------------------------------------------------
__global__ __launch_bounds__(256) void gemm_bf16_kernel(
    const u16* __restrict__ A, const u16* __restrict__ Bt,
    const float* __restrict__ bias, const float* __restrict__ resid,
    float* __restrict__ outF, u16* __restrict__ outB,
    int M, int N, int K, int relu) {
  __shared__ __align__(16) u16 As[2][128 * 32];
  __shared__ __align__(16) u16 Bs[2][128 * 32];

  const int t = threadIdx.x;
  const int lane = t & 31;
  const int wv = t >> 5;
  const int hi = lane >> 4;
  const int ln = lane & 15;
  const int wm = wv >> 1;   // 0..3
  const int wn = wv & 1;    // 0..1
  const size_t m0 = (size_t)blockIdx.y * 128;
  const size_t n0 = (size_t)blockIdx.x * 128;

  v8f acc[2][4];
#pragma unroll
  for (int i = 0; i < 2; i++)
#pragma unroll
    for (int j = 0; j < 4; j++) acc[i][j] = v8f_zero();

  const u16* Abase = A + m0 * (size_t)K;
  const u16* Bbase = Bt + n0 * (size_t)K;

#if defined(HAVE_TDM)
  // ---- prologue: DMA K-tile 0 into buffer 0
  if (wv == 0) {
    tdm_load_tile_2d(Abase, lds_off(&As[0][0]), K, 32, 128);
    tdm_load_tile_2d(Bbase, lds_off(&Bs[0][0]), K, 32, 128);
    wait_tensorcnt0();
  }
  __syncthreads();
  int cur = 0;
  for (int k0 = 0; k0 < K; k0 += 32) {
    // ---- issue DMA for next K-tile into the other buffer
    if (k0 + 32 < K && wv == 0) {
      tdm_load_tile_2d(Abase + (k0 + 32), lds_off(&As[cur ^ 1][0]), K, 32, 128);
      tdm_load_tile_2d(Bbase + (k0 + 32), lds_off(&Bs[cur ^ 1][0]), K, 32, 128);
    }
#else
  const int lrow = t >> 1;
  const int lkc = (t & 1) * 16;
  int cur = 0;
  for (int k0 = 0; k0 < K; k0 += 32) {
    *(us16*)&As[0][lrow * 32 + lkc] =
        *(const us16*)&Abase[(size_t)lrow * K + k0 + lkc];
    *(us16*)&Bs[0][lrow * 32 + lkc] =
        *(const us16*)&Bbase[(size_t)lrow * K + k0 + lkc];
    if (k0 + 32 < K) {
      __builtin_prefetch(&Abase[(size_t)lrow * K + k0 + 32 + lkc], 0, 1);
      __builtin_prefetch(&Bbase[(size_t)lrow * K + k0 + 32 + lkc], 0, 1);
    }
    __syncthreads();
#endif

    // ---- 8 WMMAs per wave from the current buffer
    us16 afr[2], bfr[4];
#pragma unroll
    for (int ms = 0; ms < 2; ms++) {
      // A-layout: lane holds row (lane&15), k = hi*16 .. +16 contiguous
      int ar = wm * 32 + ms * 16 + ln;
      us8 lo = *(const us8*)&As[cur][ar * 32 + hi * 16];
      us8 hh = *(const us8*)&As[cur][ar * 32 + hi * 16 + 8];
      afr[ms] = __builtin_shufflevector(lo, hh, 0, 1, 2, 3, 4, 5, 6, 7, 8, 9,
                                        10, 11, 12, 13, 14, 15);
    }
#pragma unroll
    for (int ns = 0; ns < 4; ns++) {
      // B-layout: lane holds column (lane&15), k = hi*16 .. +16 contiguous
      int br = wn * 64 + ns * 16 + ln;
      bfr[ns] = *(const us16*)&Bs[cur][br * 32 + hi * 16];
    }
#pragma unroll
    for (int ms = 0; ms < 2; ms++)
#pragma unroll
      for (int ns = 0; ns < 4; ns++)
        acc[ms][ns] = wmma_bf16(afr[ms], bfr[ns], acc[ms][ns]);

#if defined(HAVE_TDM)
    // next tile's DMA must be complete before the buffer flip
    if (wv == 0) wait_tensorcnt0();
    __syncthreads();
    cur ^= 1;
#else
    __syncthreads();
#endif
  }

  // epilogue: D layout = VGPR r -> row r + 8*hi, col = lane&15
#pragma unroll
  for (int ms = 0; ms < 2; ms++)
#pragma unroll
    for (int ns = 0; ns < 4; ns++)
#pragma unroll
      for (int r = 0; r < 8; r++) {
        size_t rr = m0 + wm * 32 + ms * 16 + r + 8 * hi;
        size_t cc = n0 + wn * 64 + ns * 16 + ln;
        float v = acc[ms][ns][r];
        if (bias) v += bias[cc];
        if (resid) v += resid[rr * (size_t)N + cc];
        if (relu) v = fmaxf(v, 0.f);
        if (outF) outF[rr * (size_t)N + cc] = v;
        if (outB) outB[rr * (size_t)N + cc] = f2bf(v);
      }
}

// ---------------------------------------------------------------------------
// Flash attention (DH=256). One wave per 16-row Q tile; online softmax over
// 32-key blocks; scores scaled by 1/DH (reference divides by DH, not sqrt).
// Qb,Kb: [B,S,H,DH] bf16; VT: [B,H,DH,S] bf16; Ob: [B,S,H*DH] bf16.
// ---------------------------------------------------------------------------
__global__ __launch_bounds__(256) void attn_kernel(
    const u16* __restrict__ Qb, const u16* __restrict__ Kb,
    const u16* __restrict__ VT, u16* __restrict__ Ob,
    int Bn, int S, int Hn, int causal) {
  const int DH = 256;
  __shared__ __align__(16) u16 Ps[8][16 * 32];  // per-wave P staging (D->A)

  const int t = threadIdx.x;
  const int lane = t & 31;
  const int wv = t >> 5;
  const int hi = lane >> 4;
  const int ln = lane & 15;

  const int tile = blockIdx.x * 8 + wv;      // global 16-row q-tile id
  const int qtiles = S / 16;
  const int qt = tile % qtiles;
  const int bh = tile / qtiles;
  const int h = bh % Hn;
  const int b = bh / Hn;
  const int q0 = qt * 16;
  const float scale = 1.0f / (float)DH;

  // Q fragments: 8 k-steps over DH; A-layout (row = q0+ln for both half-waves)
  us16 aq[8];
  {
    size_t qbase = (((size_t)b * S + (q0 + ln)) * Hn + h) * (size_t)DH;
#pragma unroll
    for (int ks = 0; ks < 8; ks++) {
      int d0 = ks * 32 + hi * 16;
      us8 lo = *(const us8*)(Qb + qbase + d0);
      us8 hh = *(const us8*)(Qb + qbase + d0 + 8);
      aq[ks] = __builtin_shufflevector(lo, hh, 0, 1, 2, 3, 4, 5, 6, 7, 8, 9,
                                       10, 11, 12, 13, 14, 15);
    }
  }

  float m[8], l[8], alpha[8];
#pragma unroll
  for (int r = 0; r < 8; r++) { m[r] = -3.0e38f; l[r] = 0.f; }
  v8f Oa[16];
#pragma unroll
  for (int dt = 0; dt < 16; dt++) Oa[dt] = v8f_zero();

  const size_t vbase = (((size_t)b * Hn + h) * DH) * (size_t)S;
  const int kend = causal ? (q0 + 16) : S;

  for (int kb = 0; kb < kend; kb += 32) {
    // ---- scores: Q(16x256) x K^T(256x32) -> two 16x16 fp32 tiles
    v8f s0 = v8f_zero(), s1 = v8f_zero();
#pragma unroll
    for (int ks = 0; ks < 8; ks++) {
      size_t a0 = (((size_t)b * S + (kb + ln)) * Hn + h) * (size_t)DH +
                  ks * 32 + hi * 16;
      size_t a1 = (((size_t)b * S + (kb + 16 + ln)) * Hn + h) * (size_t)DH +
                  ks * 32 + hi * 16;
      us16 bk0 = *(const us16*)(Kb + a0);
      us16 bk1 = *(const us16*)(Kb + a1);
      s0 = wmma_bf16(aq[ks], bk0, s0);
      s1 = wmma_bf16(aq[ks], bk1, s1);
    }

    // ---- online softmax (per D-layout row r+8*hi; cols across 16 lanes)
#pragma unroll
    for (int r = 0; r < 8; r++) {
      float x0 = s0[r] * scale;
      float x1 = s1[r] * scale;
      if (causal) {
        int q = q0 + r + 8 * hi;
        if (kb + ln > q) x0 = -1.0e30f;
        if (kb + 16 + ln > q) x1 = -1.0e30f;
      }
      float mx = fmaxf(x0, x1);
      mx = fmaxf(mx, __shfl_xor(mx, 1, 32));
      mx = fmaxf(mx, __shfl_xor(mx, 2, 32));
      mx = fmaxf(mx, __shfl_xor(mx, 4, 32));
      mx = fmaxf(mx, __shfl_xor(mx, 8, 32));
      float mn = fmaxf(m[r], mx);
      float al = __expf(m[r] - mn);
      float p0 = __expf(x0 - mn);
      float p1 = __expf(x1 - mn);
      float rs = p0 + p1;
      rs += __shfl_xor(rs, 1, 32);
      rs += __shfl_xor(rs, 2, 32);
      rs += __shfl_xor(rs, 4, 32);
      rs += __shfl_xor(rs, 8, 32);
      l[r] = l[r] * al + rs;
      m[r] = mn;
      alpha[r] = al;
      int prow = r + 8 * hi;
      Ps[wv][prow * 32 + ln] = f2bf(p0);
      Ps[wv][prow * 32 + 16 + ln] = f2bf(p1);
    }

    // rescale accumulator rows by alpha
#pragma unroll
    for (int dt = 0; dt < 16; dt++)
#pragma unroll
      for (int r = 0; r < 8; r++) Oa[dt][r] *= alpha[r];

    __builtin_amdgcn_wave_barrier();

    // reload P in A-operand layout from LDS
    us8 plo = *(const us8*)&Ps[wv][ln * 32 + hi * 16];
    us8 phi = *(const us8*)&Ps[wv][ln * 32 + hi * 16 + 8];
    us16 af = __builtin_shufflevector(plo, phi, 0, 1, 2, 3, 4, 5, 6, 7, 8, 9,
                                      10, 11, 12, 13, 14, 15);

    // ---- O += P(16x32) x V(32x256): 16 column tiles of 16
#pragma unroll
    for (int dt = 0; dt < 16; dt++) {
      us16 bv = *(const us16*)(VT + vbase + (size_t)(dt * 16 + ln) * S + kb +
                               hi * 16);
      Oa[dt] = wmma_bf16(af, bv, Oa[dt]);
    }
  }

  // ---- finalize: divide by l, store concat-head output [B,S,H*DH]
#pragma unroll
  for (int r = 0; r < 8; r++) l[r] = 1.0f / l[r];
#pragma unroll
  for (int dt = 0; dt < 16; dt++)
#pragma unroll
    for (int r = 0; r < 8; r++) {
      int q = q0 + r + 8 * hi;
      size_t o = (((size_t)b * S + q) * Hn + h) * (size_t)DH + dt * 16 + ln;
      Ob[o] = f2bf(Oa[dt][r] * l[r]);
    }
}

// ---------------------------------------------------------------------------
// LayerNorm over last dim E (eps = 1e-3), optional ReLU, writes f32/bf16.
// ---------------------------------------------------------------------------
__global__ __launch_bounds__(256) void ln_kernel(
    const float* __restrict__ X, const float* __restrict__ gam,
    const float* __restrict__ bet, float* __restrict__ outF,
    u16* __restrict__ outB, int E, int relu) {
  __shared__ float r1[256], r2[256];
  const int t = threadIdx.x;
  const float* xr = X + (size_t)blockIdx.x * E;
  float s = 0.f, s2 = 0.f;
  for (int e = t; e < E; e += 256) {
    float v = xr[e];
    s += v;
    s2 += v * v;
  }
  r1[t] = s;
  r2[t] = s2;
  __syncthreads();
  for (int o = 128; o > 0; o >>= 1) {
    if (t < o) {
      r1[t] += r1[t + o];
      r2[t] += r2[t + o];
    }
    __syncthreads();
  }
  float mean = r1[0] / (float)E;
  float var = r2[0] / (float)E - mean * mean;
  float rstd = rsqrtf(var + 1e-3f);
  for (int e = t; e < E; e += 256) {
    float y = (xr[e] - mean) * rstd * gam[e] + bet[e];
    if (relu) y = fmaxf(y, 0.f);
    size_t o = (size_t)blockIdx.x * E + e;
    if (outF) outF[o] = y;
    if (outB) outB[o] = f2bf(y);
  }
}

// ---------------------------------------------------------------------------
// Host orchestration
// ---------------------------------------------------------------------------
extern "C" void kernel_launch(void* const* d_in, const int* in_sizes, int n_in,
                              void* d_out, int out_size, void* d_ws,
                              size_t ws_size, hipStream_t stream) {
  (void)in_sizes; (void)n_in; (void)out_size; (void)ws_size;
  const int B = 4, S = 2048, E = 1024, H = 4, DH = 256, F = 4096;
  const size_t BS = (size_t)B * S;      // 8192
  const size_t BSE = BS * E;            // 8,388,608

  const float* inp = (const float*)d_in[0];
  const float* ctx = (const float*)d_in[1];
  const float* Wk1 = (const float*)d_in[2];  const float* bk1 = (const float*)d_in[3];
  const float* Wq1 = (const float*)d_in[4];  const float* bq1 = (const float*)d_in[5];
  const float* Wv1 = (const float*)d_in[6];  const float* bv1 = (const float*)d_in[7];
  const float* Wo1 = (const float*)d_in[8];  const float* bo1 = (const float*)d_in[9];
  const float* Wk2 = (const float*)d_in[10]; const float* bk2 = (const float*)d_in[11];
  const float* Wq2 = (const float*)d_in[12]; const float* bq2 = (const float*)d_in[13];
  const float* Wv2 = (const float*)d_in[14]; const float* bv2 = (const float*)d_in[15];
  const float* Wo2 = (const float*)d_in[16]; const float* bo2 = (const float*)d_in[17];
  const float* g_ln = (const float*)d_in[18]; const float* b_ln = (const float*)d_in[19];
  const float* W1 = (const float*)d_in[20];  const float* b1 = (const float*)d_in[21];
  const float* W2 = (const float*)d_in[22];  const float* b2 = (const float*)d_in[23];

  char* ws = (char*)d_ws;
  const size_t MB = 1ull << 20;
  u16* xb    = (u16*)(ws + 0 * MB);    // inputs bf16              16MB
  u16* cb    = (u16*)(ws + 16 * MB);   // context bf16             16MB
  u16* wk1T  = (u16*)(ws + 32 * MB);   // transposed weights (bf16)
  u16* wq1T  = (u16*)(ws + 34 * MB);
  u16* wv1T  = (u16*)(ws + 36 * MB);
  u16* wo1T  = (u16*)(ws + 38 * MB);
  u16* wk2T  = (u16*)(ws + 40 * MB);
  u16* wq2T  = (u16*)(ws + 42 * MB);
  u16* wv2T  = (u16*)(ws + 44 * MB);
  u16* wo2T  = (u16*)(ws + 46 * MB);
  u16* w1T   = (u16*)(ws + 48 * MB);   // 8MB
  u16* w2T   = (u16*)(ws + 56 * MB);   // 8MB
  u16* qbuf  = (u16*)(ws + 64 * MB);   // [B,S,H,DH]               16MB
  u16* kbuf  = (u16*)(ws + 80 * MB);   //                          16MB
  u16* vtmp  = (u16*)(ws + 96 * MB);   //                          16MB
  u16* vt    = (u16*)(ws + 112 * MB);  // [B,H,DH,S]               16MB
  u16* att   = (u16*)(ws + 128 * MB);  // attention out bf16       16MB
  float* res = (float*)(ws + 144 * MB);  // pre-LN fp32            32MB
  float* lnf = (float*)(ws + 176 * MB);  // post-LN fp32           32MB
  u16* lnb16 = (u16*)(ws + 208 * MB);    // post-LN bf16           16MB
  u16* h1    = qbuf;  // FFN hidden [8192,4096] bf16 reuses 64MB q/k/v region

  auto cg = [](size_t n) { return dim3((unsigned)((n + 255) / 256)); };

  // casts + weight transposes
  cast_bf16_kernel<<<cg(BSE), 256, 0, stream>>>(inp, xb, BSE);
  cast_bf16_kernel<<<cg(BSE), 256, 0, stream>>>(ctx, cb, BSE);
  size_t whed = (size_t)H * E * DH;
  transpose_w_kernel<<<cg(whed), 256, 0, stream>>>(Wk1, wk1T, H, E, DH);
  transpose_w_kernel<<<cg(whed), 256, 0, stream>>>(Wq1, wq1T, H, E, DH);
  transpose_w_kernel<<<cg(whed), 256, 0, stream>>>(Wv1, wv1T, H, E, DH);
  transpose_w_kernel<<<cg((size_t)E * E), 256, 0, stream>>>(Wo1, wo1T, 1, E, E);
  transpose_w_kernel<<<cg(whed), 256, 0, stream>>>(Wk2, wk2T, H, E, DH);
  transpose_w_kernel<<<cg(whed), 256, 0, stream>>>(Wq2, wq2T, H, E, DH);
  transpose_w_kernel<<<cg(whed), 256, 0, stream>>>(Wv2, wv2T, H, E, DH);
  transpose_w_kernel<<<cg((size_t)E * E), 256, 0, stream>>>(Wo2, wo2T, 1, E, E);
  transpose_w_kernel<<<cg((size_t)E * F), 256, 0, stream>>>(W1, w1T, 1, E, F);
  transpose_w_kernel<<<cg((size_t)F * E), 256, 0, stream>>>(W2, w2T, 1, F, E);

  const dim3 gProj(E / 128, (unsigned)(BS / 128));   // (8, 64)
  const dim3 gFfn1(F / 128, (unsigned)(BS / 128));   // (32, 64)
  const dim3 gAttn(256);

  // ---- self-attention (causal) ----
  gemm_bf16_kernel<<<gProj, 256, 0, stream>>>(xb, wk1T, bk1, nullptr, nullptr,
                                              kbuf, (int)BS, E, E, 0);
  gemm_bf16_kernel<<<gProj, 256, 0, stream>>>(xb, wq1T, bq1, nullptr, nullptr,
                                              qbuf, (int)BS, E, E, 0);
  gemm_bf16_kernel<<<gProj, 256, 0, stream>>>(xb, wv1T, bv1, nullptr, nullptr,
                                              vtmp, (int)BS, E, E, 0);
  transpose_v_kernel<<<cg(BSE), 256, 0, stream>>>(vtmp, vt, B, S, H, DH);
  attn_kernel<<<gAttn, 256, 0, stream>>>(qbuf, kbuf, vt, att, B, S, H, 1);
  gemm_bf16_kernel<<<gProj, 256, 0, stream>>>(att, wo1T, bo1, inp, res,
                                              nullptr, (int)BS, E, E, 0);
  ln_kernel<<<dim3((unsigned)BS), 256, 0, stream>>>(res, g_ln, b_ln, lnf,
                                                    lnb16, E, 0);

  // ---- cross-attention (no mask); Q from normalized a, K/V from context ----
  gemm_bf16_kernel<<<gProj, 256, 0, stream>>>(cb, wk2T, bk2, nullptr, nullptr,
                                              kbuf, (int)BS, E, E, 0);
  gemm_bf16_kernel<<<gProj, 256, 0, stream>>>(lnb16, wq2T, bq2, nullptr,
                                              nullptr, qbuf, (int)BS, E, E, 0);
  gemm_bf16_kernel<<<gProj, 256, 0, stream>>>(cb, wv2T, bv2, nullptr, nullptr,
                                              vtmp, (int)BS, E, E, 0);
  transpose_v_kernel<<<cg(BSE), 256, 0, stream>>>(vtmp, vt, B, S, H, DH);
  attn_kernel<<<gAttn, 256, 0, stream>>>(qbuf, kbuf, vt, att, B, S, H, 0);
  gemm_bf16_kernel<<<gProj, 256, 0, stream>>>(att, wo2T, bo2, lnf, res,
                                              nullptr, (int)BS, E, E, 0);
  ln_kernel<<<dim3((unsigned)BS), 256, 0, stream>>>(res, g_ln, b_ln, lnf,
                                                    lnb16, E, 0);

  // ---- FFN ----
  gemm_bf16_kernel<<<gFfn1, 256, 0, stream>>>(lnb16, w1T, b1, nullptr, nullptr,
                                              h1, (int)BS, F, E, 1);
  gemm_bf16_kernel<<<gProj, 256, 0, stream>>>(h1, w2T, b2, lnf, res, nullptr,
                                              (int)BS, E, F, 0);
  ln_kernel<<<dim3((unsigned)BS), 256, 0, stream>>>(res, g_ln, b_ln,
                                                    (float*)d_out, nullptr, E,
                                                    1);
}
